// MoleculeModel_45792941310672
// MI455X (gfx1250) — matrix operand order repaired and checked by reading, not compiled, not run on hardware
//
#include <hip/hip_runtime.h>

#define GENES 978
#define HIDG  300
#define BATCH 128
#define FD    512
#define GT    62      // ceil(978/16) gene tiles
#define CHUNKS 8

typedef __attribute__((ext_vector_type(16))) __bf16 bf16x16;
typedef __attribute__((ext_vector_type(8)))  __bf16 bf16x8;
typedef __attribute__((ext_vector_type(8)))  float  f32x8;
typedef __attribute__((ext_vector_type(4)))  float  f32x4;

// ---------------- small fp32 helper kernels ----------------

__global__ void k_zero(float* __restrict__ p, int n) {
    int i = blockIdx.x * blockDim.x + threadIdx.x;
    if (i < n) p[i] = 0.f;
}

// in-degree (incl. self loops). edge_index flat [2,E]: [0..E)=src, [E..2E)=dst
__global__ void k_deg(const int* __restrict__ eidx, int E, float* __restrict__ deg) {
    int i = blockIdx.x * blockDim.x + threadIdx.x;
    if (i < E + GENES) {
        int d = (i < E) ? eidx[E + i] : (i - E);
        atomicAdd(&deg[d], 1.0f);
    }
}

__global__ void k_dinv(const float* __restrict__ deg, float* __restrict__ dinv) {
    int i = blockIdx.x * blockDim.x + threadIdx.x;
    if (i < GENES) {
        float d = deg[i];
        dinv[i] = d > 0.f ? rsqrtf(d) : 0.f;
    }
}

// C[M,N] = A[M,K] @ W[K,N]   (naive fp32; only used for tiny matmuls)
__global__ void k_mm(const float* __restrict__ A, const float* __restrict__ W,
                     float* __restrict__ C, int M, int K, int N) {
    int i = blockIdx.x * blockDim.x + threadIdx.x;
    if (i >= M * N) return;
    int m = i / N, n = i % N;
    float s = 0.f;
    for (int k = 0; k < K; ++k) s += A[m * K + k] * W[k * N + n];
    C[i] = s;
}

// out[dst] += h[src] * dinv[src]*dinv[dst]  over E edges + self loops
__global__ void k_scatter(const int* __restrict__ eidx, int E,
                          const float* __restrict__ h, const float* __restrict__ dinv,
                          float* __restrict__ out, int D) {
    int e = blockIdx.x;
    if (e >= E + GENES) return;
    int s, d;
    if (e < E) { s = eidx[e]; d = eidx[E + e]; } else { s = e - E; d = s; }
    float norm = dinv[s] * dinv[d];
    for (int j = threadIdx.x; j < D; j += blockDim.x)
        atomicAdd(&out[d * D + j], h[s * D + j] * norm);
}

__global__ void k_addbias(float* __restrict__ x, const float* __restrict__ b,
                          int total, int N) {
    int i = blockIdx.x * blockDim.x + threadIdx.x;
    if (i < total) x[i] += b[i % N];
}

// pack 512x512 fp32 row-major W[k][n] into bf16 WMMA B-fragment layout:
// Wp[(((ntile*16 + kb)*32 + lane)*16 + j)] = W[kb*32 + (j>>3)*16 + (lane>>4)*8 + (j&7)][ntile*16 + (lane&15)]
__global__ void k_pack(const float* __restrict__ W, __bf16* __restrict__ Wp) {
    int i = blockIdx.x * blockDim.x + threadIdx.x;
    if (i >= FD * FD) return;
    int j     = i & 15;
    int lane  = (i >> 4) & 31;
    int kb    = (i >> 9) & 15;
    int ntile = i >> 13;
    int n = ntile * 16 + (lane & 15);
    int k = kb * 32 + (j >> 3) * 16 + (lane >> 4) * 8 + (j & 7);
    Wp[i] = (__bf16)W[k * FD + n];
}

// out[b][j] = b4[j] + (1/978) * sum_n acc[b][n] * W4[n][j]
__global__ void k_out(const float* __restrict__ acc, const float* __restrict__ W4,
                      const float* __restrict__ b4, float* __restrict__ out) {
    int i = blockIdx.x * blockDim.x + threadIdx.x;
    if (i >= BATCH * GENES) return;
    int b = i / GENES, j = i % GENES;
    float s = 0.f;
    for (int n = 0; n < FD; ++n) s += acc[b * FD + n] * W4[n * GENES + j];
    out[i] = b4[j] + s * (1.0f / (float)GENES);
}

// ---------------- fused bf16 WMMA chain ----------------

__device__ inline bf16x16 make_xfrag(const f32x4 hg[4], const f32x4 hm[4]) {
    bf16x16 a;
#pragma unroll
    for (int q = 0; q < 4; ++q) {
#pragma unroll
        for (int j = 0; j < 4; ++j) {
            float v = hg[q][j] + hm[q][j];
            v = v > 0.f ? v : 0.f;
            a[q * 4 + j] = (__bf16)v;
        }
    }
    return a;
}

__device__ inline bf16x16 load_yfrag(const __bf16* base, int k0, int hi8) {
    bf16x8 lo = *(const bf16x8*)(base + k0 + hi8);
    bf16x8 hi = *(const bf16x8*)(base + k0 + 16 + hi8);
    return __builtin_shufflevector(lo, hi, 0,1,2,3,4,5,6,7,8,9,10,11,12,13,14,15);
}

__global__ __launch_bounds__(256) void k_fused(
    const float*  __restrict__ hg2,  // [978][512] fp32, ffn1_b folded in
    const float*  __restrict__ hm,   // [128][512] fp32
    const __bf16* __restrict__ W2p,  // packed B fragments
    const float*  __restrict__ b2,
    const __bf16* __restrict__ W3p,
    const float*  __restrict__ b3,
    float*        __restrict__ acc)  // [128][512], pre-zeroed; sum over genes
{
    __shared__ __bf16 yLDS[2][16][FD];   // 32 KB

    const int pair  = blockIdx.x / CHUNKS;     // 0..63 (batch pair)
    const int chunk = blockIdx.x % CHUNKS;
    const int b0    = pair * 2;
    const int wave  = threadIdx.x >> 5;        // 0..7
    const int lane  = threadIdx.x & 31;
    const int ln    = lane & 15;
    const int hi8   = (lane >> 4) * 8;

    const float* hm0 = hm + (size_t)b0 * FD;
    const float* hm1 = hm0 + FD;

    float sum0[4], sum1[4];
#pragma unroll
    for (int nt = 0; nt < 4; ++nt) { sum0[nt] = 0.f; sum1[nt] = 0.f; }

    const f32x8 vzero = {0,0,0,0,0,0,0,0};

    for (int t = chunk; t < GT; t += CHUNKS) {
        const int g0 = t * 16;
        int g  = g0 + ln;
        int gg = g < GENES ? g : (GENES - 1);        // clamp; masked later
        const float* hgrow = hg2 + (size_t)gg * FD;

        // ---------- layer ffn2: y = relu(x @ W2 + b2), x generated on the fly ----------
        f32x8 c0[4], c1[4];
#pragma unroll
        for (int nt = 0; nt < 4; ++nt) { c0[nt] = vzero; c1[nt] = vzero; }

        for (int kb = 0; kb < 16; ++kb) {
            const int k0 = kb * 32;
            f32x4 hgv[4], hv0[4], hv1[4];
            const int o[4] = { k0 + hi8, k0 + hi8 + 4, k0 + 16 + hi8, k0 + 16 + hi8 + 4 };
#pragma unroll
            for (int q = 0; q < 4; ++q) {
                hgv[q] = *(const f32x4*)(hgrow + o[q]);
                hv0[q] = *(const f32x4*)(hm0 + o[q]);
                hv1[q] = *(const f32x4*)(hm1 + o[q]);
            }
            bf16x16 A0 = make_xfrag(hgv, hv0);
            bf16x16 A1 = make_xfrag(hgv, hv1);

            if (kb + 1 < 16)   // pull next K-block of W2 toward the WGP
                __builtin_prefetch(W2p + (((size_t)(wave * 4) * 16 + (kb + 1)) * 32 + lane) * 16, 0, 0);

#pragma unroll
            for (int nt = 0; nt < 4; ++nt) {
                const int ntile = wave * 4 + nt;
                bf16x16 B = *(const bf16x16*)(W2p + (((size_t)ntile * 16 + kb) * 32 + lane) * 16);
                c0[nt] = __builtin_amdgcn_wmma_f32_16x16x32_bf16(false, A0, false, B, (short)0, c0[nt], false, false);
                c1[nt] = __builtin_amdgcn_wmma_f32_16x16x32_bf16(false, A1, false, B, (short)0, c1[nt], false, false);
            }
        }
        // epilogue: bias + relu -> yLDS (bf16)
#pragma unroll
        for (int nt = 0; nt < 4; ++nt) {
            const int n = wave * 64 + nt * 16 + ln;
            const float bv = b2[n];
#pragma unroll
            for (int r = 0; r < 8; ++r) {
                const int row = hi8 + r;
                float v0 = c0[nt][r] + bv; v0 = v0 > 0.f ? v0 : 0.f;
                float v1 = c1[nt][r] + bv; v1 = v1 > 0.f ? v1 : 0.f;
                yLDS[0][row][n] = (__bf16)v0;
                yLDS[1][row][n] = (__bf16)v1;
            }
        }
        __syncthreads();

        // ---------- layer ffn3: z = relu(y @ W3 + b3); reduce z over genes ----------
        f32x8 d0[4], d1[4];
#pragma unroll
        for (int nt = 0; nt < 4; ++nt) { d0[nt] = vzero; d1[nt] = vzero; }

        for (int kb = 0; kb < 16; ++kb) {
            const int k0 = kb * 32;
            bf16x16 A0 = load_yfrag(&yLDS[0][ln][0], k0, hi8);
            bf16x16 A1 = load_yfrag(&yLDS[1][ln][0], k0, hi8);
#pragma unroll
            for (int nt = 0; nt < 4; ++nt) {
                const int ntile = wave * 4 + nt;
                bf16x16 B = *(const bf16x16*)(W3p + (((size_t)ntile * 16 + kb) * 32 + lane) * 16);
                d0[nt] = __builtin_amdgcn_wmma_f32_16x16x32_bf16(false, A0, false, B, (short)0, d0[nt], false, false);
                d1[nt] = __builtin_amdgcn_wmma_f32_16x16x32_bf16(false, A1, false, B, (short)0, d1[nt], false, false);
            }
        }
        // epilogue: bias + relu, masked row-sum into register accumulators
#pragma unroll
        for (int nt = 0; nt < 4; ++nt) {
            const int n = wave * 64 + nt * 16 + ln;
            const float bv = b3[n];
            float s0 = 0.f, s1 = 0.f;
#pragma unroll
            for (int r = 0; r < 8; ++r) {
                const int row = hi8 + r;
                const float mask = (g0 + row) < GENES ? 1.f : 0.f;
                float v0 = d0[nt][r] + bv; v0 = v0 > 0.f ? v0 : 0.f;
                float v1 = d1[nt][r] + bv; v1 = v1 > 0.f ? v1 : 0.f;
                s0 += v0 * mask;
                s1 += v1 * mask;
            }
            sum0[nt] += s0;
            sum1[nt] += s1;
        }
        __syncthreads();   // protect yLDS before next tile overwrites it
    }

    // flush per-lane partial gene-sums (lane l and l+16 hit the same column -> atomics)
#pragma unroll
    for (int nt = 0; nt < 4; ++nt) {
        const int n = wave * 64 + nt * 16 + ln;
        atomicAdd(&acc[(size_t)b0 * FD + n], sum0[nt]);
        atomicAdd(&acc[(size_t)(b0 + 1) * FD + n], sum1[nt]);
    }
}

// ---------------- host driver ----------------

extern "C" void kernel_launch(void* const* d_in, const int* in_sizes, int n_in,
                              void* d_out, int out_size, void* d_ws, size_t ws_size,
                              hipStream_t stream) {
    const float* mol  = (const float*)d_in[0];
    const float* gemb = (const float*)d_in[1];
    const int*   eidx = (const int*)  d_in[2];
    const float* w1   = (const float*)d_in[3];
    const float* b1g  = (const float*)d_in[4];
    const float* w2g  = (const float*)d_in[5];
    const float* b2g  = (const float*)d_in[6];
    const float* fw1  = (const float*)d_in[7];
    const float* fb1  = (const float*)d_in[8];
    const float* fw2  = (const float*)d_in[9];
    const float* fb2  = (const float*)d_in[10];
    const float* fw3  = (const float*)d_in[11];
    const float* fb3  = (const float*)d_in[12];
    const float* fw4  = (const float*)d_in[13];
    const float* fb4  = (const float*)d_in[14];
    const int E = in_sizes[2] / 2;

    char* ws = (char*)d_ws;
    size_t off = 0;
    auto alloc = [&](size_t bytes) -> void* {
        void* p = ws + off;
        off = (off + bytes + 255) & ~(size_t)255;
        return p;
    };
    float*  deg  = (float*) alloc((size_t)GENES * 4);
    float*  dinv = (float*) alloc((size_t)GENES * 4);
    float*  htmp = (float*) alloc((size_t)GENES * HIDG * 4);
    float*  g1   = (float*) alloc((size_t)GENES * HIDG * 4);
    float*  g2   = (float*) alloc((size_t)GENES * HIDG * 4);
    float*  hg2  = (float*) alloc((size_t)GENES * FD * 4);
    float*  hmb  = (float*) alloc((size_t)BATCH * FD * 4);
    float*  acc  = (float*) alloc((size_t)BATCH * FD * 4);
    __bf16* W2p  = (__bf16*)alloc((size_t)FD * FD * 2);
    __bf16* W3p  = (__bf16*)alloc((size_t)FD * FD * 2);

    auto blocks = [](int n) { return dim3((unsigned)((n + 255) / 256)); };

    // zero accumulation targets (re-zeroed every call: graph replays reuse ws)
    k_zero<<<blocks(GENES), 256, 0, stream>>>(deg, GENES);
    k_zero<<<blocks(GENES * HIDG), 256, 0, stream>>>(g1, GENES * HIDG);
    k_zero<<<blocks(GENES * HIDG), 256, 0, stream>>>(g2, GENES * HIDG);
    k_zero<<<blocks(BATCH * FD), 256, 0, stream>>>(acc, BATCH * FD);

    // GCN graph normalization
    k_deg <<<blocks(E + GENES), 256, 0, stream>>>(eidx, E, deg);
    k_dinv<<<blocks(GENES),     256, 0, stream>>>(deg, dinv);

    // GCN layer 1
    k_mm     <<<blocks(GENES * HIDG), 256, 0, stream>>>(gemb, w1, htmp, GENES, HIDG, HIDG);
    k_scatter<<<dim3((unsigned)(E + GENES)), 256, 0, stream>>>(eidx, E, htmp, dinv, g1, HIDG);
    k_addbias<<<blocks(GENES * HIDG), 256, 0, stream>>>(g1, b1g, GENES * HIDG, HIDG);

    // GCN layer 2
    k_mm     <<<blocks(GENES * HIDG), 256, 0, stream>>>(g1, w2g, htmp, GENES, HIDG, HIDG);
    k_scatter<<<dim3((unsigned)(E + GENES)), 256, 0, stream>>>(eidx, E, htmp, dinv, g2, HIDG);
    k_addbias<<<blocks(GENES * HIDG), 256, 0, stream>>>(g2, b2g, GENES * HIDG, HIDG);

    // ffn1 split matmuls; fold ffn1_b into hg2
    k_mm     <<<blocks(GENES * FD), 256, 0, stream>>>(g2, fw1, hg2, GENES, HIDG, FD);
    k_addbias<<<blocks(GENES * FD), 256, 0, stream>>>(hg2, fb1, GENES * FD, FD);
    k_mm     <<<blocks(BATCH * FD), 256, 0, stream>>>(mol, fw1 + (size_t)HIDG * FD, hmb, BATCH, HIDG, FD);

    // pack ffn2/ffn3 weights into bf16 WMMA B-fragment layout
    k_pack<<<blocks(FD * FD), 256, 0, stream>>>(fw2, W2p);
    k_pack<<<blocks(FD * FD), 256, 0, stream>>>(fw3, W3p);

    // fused ffn2+ffn3+gene-mean (the 131-GFLOP core, all bf16 WMMA)
    k_fused<<<dim3((unsigned)((BATCH / 2) * CHUNKS)), 256, 0, stream>>>(
        hg2, hmb, W2p, fb2, W3p, fb3, acc);

    // final projection to 978 genes
    k_out<<<blocks(BATCH * GENES), 256, 0, stream>>>(acc, fw4, fb4, (float*)d_out);
}
